// Closedbook_77790447665286
// MI455X (gfx1250) — compile-verified, matrix operand order checked by reading, not run on hardware
//
#include <hip/hip_runtime.h>
#include <hip/hip_fp16.h>

// Problem constants (reference: B,H,TK,V,NOOV = 512,256,512,50000,64)
#define B_    512
#define H_    256
#define TK_   512
#define V_    50000
#define NOOV_ 64
#define VE_   (V_ + NOOV_)   // 50064 extended-vocab row stride

typedef _Float16 f16;
typedef __attribute__((ext_vector_type(16))) _Float16 v16h;
typedef __attribute__((ext_vector_type(8)))  _Float16 v8h;
typedef __attribute__((ext_vector_type(4)))  _Float16 v4h;
typedef __attribute__((ext_vector_type(8)))  float    v8f;
typedef __attribute__((ext_vector_type(4)))  unsigned int u32x4;
typedef __attribute__((ext_vector_type(4)))  int          i32x4;
typedef __attribute__((ext_vector_type(8)))  int          i32x8;

// ---------------- reduction helpers (wave32) ----------------
__device__ __forceinline__ float wave_sum(float v) {
  #pragma unroll
  for (int off = 16; off; off >>= 1) v += __shfl_down(v, off, 32);
  return v;
}
__device__ __forceinline__ float wave_max(float v) {
  #pragma unroll
  for (int off = 16; off; off >>= 1) v = fmaxf(v, __shfl_down(v, off, 32));
  return v;
}
// 256-thread block reductions; red must hold >= 8 floats
__device__ __forceinline__ float block_sum(float v, float* red, int tid) {
  v = wave_sum(v);
  if ((tid & 31) == 0) red[tid >> 5] = v;
  __syncthreads();
  float r = red[0];
  #pragma unroll
  for (int w = 1; w < 8; ++w) r += red[w];
  __syncthreads();
  return r;
}
__device__ __forceinline__ float block_max(float v, float* red, int tid) {
  v = wave_max(v);
  if ((tid & 31) == 0) red[tid >> 5] = v;
  __syncthreads();
  float r = red[0];
  #pragma unroll
  for (int w = 1; w < 8; ++w) r = fmaxf(r, red[w]);
  __syncthreads();
  return r;
}
__device__ __forceinline__ float sigmoidf_(float x) { return 1.f / (1.f + __expf(-x)); }
__device__ __forceinline__ float fast_tanh(float x) {
  // 1 - 2/(e^{2x}+1): saturates correctly at +-1, one v_exp + one rcp
  float e2 = __expf(2.f * x);
  return 1.f - 2.f / (e2 + 1.f);
}

// ---------------- K1: x = concat(c_t_1, emb[y]) @ Wx^T + bx ----------------
__global__ void __launch_bounds__(256)
k_proj_x(const float* __restrict__ c_t_1, const float* __restrict__ emb,
         const int* __restrict__ y, const float* __restrict__ Wx,
         const float* __restrict__ bx, float* __restrict__ x) {
  __shared__ float cat[3 * H_];
  int b = blockIdx.x, tid = threadIdx.x;
  cat[tid]        = c_t_1[b * 2 * H_ + tid];
  cat[256 + tid]  = c_t_1[b * 2 * H_ + 256 + tid];
  int yb = y[b];
  cat[512 + tid]  = emb[(size_t)yb * H_ + tid];
  __syncthreads();
  const float* w = Wx + tid * (3 * H_);
  float acc = bx[tid];
  #pragma unroll 4
  for (int k = 0; k < 3 * H_; ++k) acc = fmaf(cat[k], w[k], acc);
  x[b * H_ + tid] = acc;
}

// ---------------- K2: LSTM cell (gate order i,f,g,o) ----------------
__global__ void __launch_bounds__(256)
k_lstm(const float* __restrict__ x, const float* __restrict__ h0,
       const float* __restrict__ c0,
       const float* __restrict__ W_ih, const float* __restrict__ W_hh,
       const float* __restrict__ b_ih, const float* __restrict__ b_hh,
       float* __restrict__ h_out, float* __restrict__ c_out) {
  __shared__ float xs[H_], hs[H_], gates[4 * H_];
  int b = blockIdx.x, tid = threadIdx.x;
  xs[tid] = x[b * H_ + tid];
  hs[tid] = h0[b * H_ + tid];
  __syncthreads();
  #pragma unroll
  for (int g = 0; g < 4; ++g) {
    int j = g * H_ + tid;
    const float* wi = W_ih + j * H_;
    const float* wh = W_hh + j * H_;
    float acc = b_ih[j] + b_hh[j];
    #pragma unroll 4
    for (int k = 0; k < H_; ++k) acc = fmaf(xs[k], wi[k], fmaf(hs[k], wh[k], acc));
    gates[j] = acc;
  }
  __syncthreads();
  float gi = gates[tid], gf = gates[H_ + tid], gg = gates[2 * H_ + tid], go = gates[3 * H_ + tid];
  float c = sigmoidf_(gf) * c0[b * H_ + tid] + sigmoidf_(gi) * fast_tanh(gg);
  float h = sigmoidf_(go) * fast_tanh(c);
  h_out[b * H_ + tid] = h;
  c_out[b * H_ + tid] = c;
}

// ---------------- K3: dec_fea = [h|c] @ Wdp^T + bdp ----------------
__global__ void __launch_bounds__(256)
k_decfea(const float* __restrict__ h, const float* __restrict__ c,
         const float* __restrict__ Wdp, const float* __restrict__ bdp,
         float* __restrict__ dec_fea) {
  __shared__ float s[2 * H_];
  int b = blockIdx.x, tid = threadIdx.x;
  s[tid]       = h[b * H_ + tid];
  s[H_ + tid]  = c[b * H_ + tid];
  __syncthreads();
  #pragma unroll
  for (int jj = 0; jj < 2; ++jj) {
    int j = tid + jj * 256;
    const float* w = Wdp + j * (2 * H_);
    float acc = bdp[j];
    #pragma unroll 4
    for (int k = 0; k < 2 * H_; ++k) acc = fmaf(s[k], w[k], acc);
    dec_fea[b * 2 * H_ + j] = acc;
  }
}

// ---------------- K4: fused attention: scores -> softmax -> c_t ----------------
// Never materializes e[B,TK,2H] (would be 512 MB).
__global__ void __launch_bounds__(256)
k_attn(const float* __restrict__ dec_fea, const float* __restrict__ mem,
       const float* __restrict__ v_w,
       float* __restrict__ attn_out, float* __restrict__ ct_out) {
  __shared__ float df[2 * H_], vw[2 * H_], sc[TK_], red[8];
  int b = blockIdx.x, tid = threadIdx.x;
  df[tid]       = dec_fea[b * 2 * H_ + tid];
  df[256 + tid] = dec_fea[b * 2 * H_ + 256 + tid];
  vw[tid]       = v_w[tid];
  vw[256 + tid] = v_w[256 + tid];
  __syncthreads();
  int wave = tid >> 5, lane = tid & 31;
  // scores: one wave per key t, lanes split the 2H reduction (coalesced mem reads)
  for (int t = wave; t < TK_; t += 8) {
    const float* m = mem + (size_t)t * (2 * H_);
    float acc = 0.f;
    #pragma unroll 4
    for (int d = lane; d < 2 * H_; d += 32)
      acc = fmaf(vw[d], fast_tanh(m[d] + df[d]), acc);
    acc = wave_sum(acc);
    if (lane == 0) sc[t] = acc;
  }
  __syncthreads();
  // softmax over TK (renormalization in the reference is a mathematical no-op)
  float p0 = sc[tid], p1 = sc[tid + 256];
  float mx = block_max(fmaxf(p0, p1), red, tid);
  float e0 = __expf(p0 - mx), e1 = __expf(p1 - mx);
  float sum = block_sum(e0 + e1, red, tid);
  float inv = 1.f / sum;
  sc[tid]       = e0 * inv;
  sc[tid + 256] = e1 * inv;
  attn_out[b * TK_ + tid]       = e0 * inv;
  attn_out[b * TK_ + tid + 256] = e1 * inv;
  __syncthreads();
  // c_t = attn @ mem  (each thread owns 2 columns; coalesced along d)
  float a0 = 0.f, a1 = 0.f;
  for (int t = 0; t < TK_; ++t) {
    float at = sc[t];
    const float* m = mem + (size_t)t * (2 * H_);
    a0 = fmaf(at, m[tid], a0);
    a1 = fmaf(at, m[tid + 256], a1);
  }
  ct_out[b * 2 * H_ + tid]       = a0;
  ct_out[b * 2 * H_ + tid + 256] = a1;
}

// ---------------- K5: p_gen + out1 = relu([h|c_t]@Wo1^T+bo1), f16 convert ----------------
__global__ void __launch_bounds__(256)
k_out1_pgen(const float* __restrict__ h, const float* __restrict__ c,
            const float* __restrict__ ct, const float* __restrict__ x,
            const float* __restrict__ Wpg, const float* __restrict__ bpg,
            const float* __restrict__ Wo1, const float* __restrict__ bo1,
            float* __restrict__ pgen_out, f16* __restrict__ out1h) {
  __shared__ float zz[5 * H_];   // [c_t(512) | h(256) | c(256) | x(256)]
  __shared__ float red[8];
  int b = blockIdx.x, tid = threadIdx.x;
  zz[tid]         = ct[b * 2 * H_ + tid];
  zz[256 + tid]   = ct[b * 2 * H_ + 256 + tid];
  zz[512 + tid]   = h[b * H_ + tid];
  zz[768 + tid]   = c[b * H_ + tid];
  zz[1024 + tid]  = x[b * H_ + tid];
  __syncthreads();
  float part = 0.f;
  #pragma unroll
  for (int j = 0; j < 5; ++j)
    part = fmaf(zz[tid + j * 256], Wpg[tid + j * 256], part);
  float tot = block_sum(part, red, tid);
  float pg = sigmoidf_(tot + bpg[0]);
  if (tid == 0) pgen_out[b] = pg;
  // out1 row: cat2 = [h(256) | c_t(512)]
  const float* w = Wo1 + tid * (3 * H_);
  float acc = bo1[tid];
  #pragma unroll 4
  for (int k = 0; k < 3 * H_; ++k) {
    float v = (k < H_) ? zz[512 + k] : zz[k - H_];
    acc = fmaf(v, w[k], acc);
  }
  out1h[b * H_ + tid] = (f16)fmaxf(acc, 0.f);
}

// ---------------- K6: Wo2 fp32 -> f16 (streaming, prefetched) ----------------
__global__ void __launch_bounds__(256)
k_cvt_w(const float* __restrict__ w, f16* __restrict__ wh, int n) {
  int i = (blockIdx.x * 256 + threadIdx.x) * 4;
  if (i >= n) return;
  __builtin_prefetch(w + i + 4096, 0, 0);     // global_prefetch_b8
  float4 v = *(const float4*)(w + i);
  v4h o;
  o[0] = (f16)v.x; o[1] = (f16)v.y; o[2] = (f16)v.z; o[3] = (f16)v.w;
  *(v4h*)(wh + i) = o;
}

// ---------------- K7: vocab GEMM via v_wmma_f32_16x16x32_f16 + TDM-staged A ----------------
// logits[512,50000] = out1h[512,256] @ Wo2h^T + bo2, written with stride VE_
// Block = 8 waves. blockIdx.x selects an M-group of 4 tiles (64 contiguous A rows,
// 32 KB) staged once into LDS via the Tensor Data Mover and shared by all waves.
// Each wave owns one 16-col N tile, holds its full B fragment set (8 x v16h = 64
// VGPRs) in registers and reuses it across the 4 M-tiles -> 32 WMMAs per wave.
// Fragment layout per CDNA5 ISA (16-bit A 16x32): lane l (half=l>>4, q=l&15)
// holds K chunks {k0+half*8..+7} and {k0+16+half*8..+7}; D: col q, rows r+8*half.
#define MTILES_PER_BLK 4
__global__ void __launch_bounds__(256)
k_gemm_vocab(const f16* __restrict__ A, const f16* __restrict__ Bw,
             const float* __restrict__ bo2, float* __restrict__ out) {
  __shared__ __align__(16) f16 Ash[MTILES_PER_BLK * 16 * H_];   // 32 KB
  int tid   = threadIdx.x;
  int wave  = tid >> 5;
  int lane  = tid & 31;
  int mg    = blockIdx.x;                 // 0..7   (M = 8 * 64)
  int ntile = blockIdx.y * 8 + wave;      // 0..3124 (N = 3125 * 16)

  // ---- stage 64 contiguous A rows (32 KB) into LDS ----
  const f16* asrc = A + (size_t)mg * (MTILES_PER_BLK * 16) * H_;
#if defined(__HIP_DEVICE_COMPILE__) && __has_builtin(__builtin_amdgcn_tensor_load_to_lds)
  if (wave == 0) {
    unsigned long long ga = (unsigned long long)(const void*)asrc;
    unsigned int lds_lo   = (unsigned int)(size_t)&Ash[0];   // LDS offset = addr[31:0]
    u32x4 g0;
    g0[0] = 1u;                                              // count=1, user D#
    g0[1] = lds_lo;                                          // lds_addr
    g0[2] = (unsigned int)ga;                                // global_addr[31:0]
    g0[3] = (unsigned int)((ga >> 32) & 0x01FFFFFFu) | (2u << 30);  // addr[56:32] | type=2
    i32x8 g1;
    g1[0] = (int)(1u << 16);                 // workgroup_mask=0, data_size=1 (2 B)
    g1[1] = (int)((16384u & 0xFFFFu) << 16); // tensor_dim0[15:0]=16384
    g1[2] = (int)(16384u >> 16);             // tensor_dim0[31:16] | tensor_dim1=0
    g1[3] = (int)(16384u << 16);             // tile_dim0 = 16384 elements (1-D copy)
    g1[4] = 0;                               // tile_dim1=0 (unused), tile_dim2=0
    g1[5] = (int)16384u;                     // tensor_dim0_stride
    g1[6] = 0; g1[7] = 0;
    i32x4 zero4 = {0, 0, 0, 0};
#if __clang_major__ >= 23
    i32x8 zero8 = {0, 0, 0, 0, 0, 0, 0, 0};
    __builtin_amdgcn_tensor_load_to_lds(g0, g1, zero4, zero4, zero8, 0);
#else
    __builtin_amdgcn_tensor_load_to_lds(g0, g1, zero4, zero4, 0);
#endif
    __builtin_amdgcn_s_wait_tensorcnt(0);
  }
#else
  {
    const uint4* s = (const uint4*)asrc;
    uint4* d = (uint4*)&Ash[0];
    #pragma unroll
    for (int j = tid; j < (MTILES_PER_BLK * 16 * H_) / 8; j += 256) d[j] = s[j];
  }
#endif
  __syncthreads();

  if (ntile < 3125) {                       // wave-uniform: EXEC all-ones inside
    int half = lane >> 4;
    int q    = lane & 15;
    // ---- load full B fragment set once (reused across 4 M-tiles) ----
    const f16* brow = Bw + (size_t)(ntile * 16 + q) * H_;
    v16h bfr[8];
    #pragma unroll
    for (int kk = 0; kk < 8; ++kk) {
      int k0 = kk * 32 + half * 8;
      v8h b0 = *(const v8h*)(brow + k0);
      v8h b1 = *(const v8h*)(brow + k0 + 16);
      bfr[kk] = __builtin_shufflevector(b0, b1, 0,1,2,3,4,5,6,7,8,9,10,11,12,13,14,15);
    }
    int col = ntile * 16 + q;               // < 50000 by construction
    float bias = bo2[col];
    #pragma unroll
    for (int m = 0; m < MTILES_PER_BLK; ++m) {
      const f16* arow = &Ash[(m * 16 + q) * H_];
      v8f acc = {};
      #pragma unroll
      for (int kk = 0; kk < 8; ++kk) {      // K = 8 * 32, A from LDS (ds_load_b128)
        int k0 = kk * 32 + half * 8;
        v8h a0 = *(const v8h*)(arow + k0);
        v8h a1 = *(const v8h*)(arow + k0 + 16);
        v16h a = __builtin_shufflevector(a0, a1, 0,1,2,3,4,5,6,7,8,9,10,11,12,13,14,15);
        acc = __builtin_amdgcn_wmma_f32_16x16x32_f16(false, a, false, bfr[kk],
                                                     (short)0, acc, false, false);
      }
      float* orow = out + (size_t)(mg * 64 + m * 16 + 8 * half) * VE_ + col;
      #pragma unroll
      for (int r = 0; r < 8; ++r) orow[(size_t)r * VE_] = acc[r] + bias;
    }
  }
}

// ---------------- K8: row softmax over V, scale by p_gen, zero OOV tail ----------------
__global__ void __launch_bounds__(256)
k_softmax(float* __restrict__ fd, const float* __restrict__ pgen) {
  __shared__ float red[8];
  int b = blockIdx.x, tid = threadIdx.x;
  float* row = fd + (size_t)b * VE_;
  float m = -3.4e38f;
  for (int j = tid; j < V_; j += 256) m = fmaxf(m, row[j]);
  float mx = block_max(m, red, tid);
  float s = 0.f;
  for (int j = tid; j < V_; j += 256) {
    float e = __expf(row[j] - mx);
    row[j] = e;                      // store exp once; scaled in next pass
    s += e;
  }
  float sum = block_sum(s, red, tid);
  float scale = pgen[b] / sum;
  for (int j = tid; j < V_; j += 256) row[j] *= scale;
  for (int j = V_ + tid; j < VE_; j += 256) row[j] = 0.f;
}

// ---------------- K9: pointer scatter-add + coverage pass-through ----------------
__global__ void __launch_bounds__(256)
k_scatter(float* __restrict__ fd, const float* __restrict__ attn,
          const float* __restrict__ pgen, const int* __restrict__ ebv,
          const float* __restrict__ cov_in, float* __restrict__ cov_out) {
  int i = blockIdx.x * 256 + threadIdx.x;      // 0 .. B*TK-1
  if (i >= B_ * TK_) return;
  int b = i >> 9;                               // TK_ == 512
  float ad = (1.f - pgen[b]) * attn[i];
  atomicAdd(fd + (size_t)b * VE_ + ebv[i], ad);
  cov_out[i] = cov_in[i];
}

extern "C" void kernel_launch(void* const* d_in, const int* in_sizes, int n_in,
                              void* d_out, int out_size, void* d_ws, size_t ws_size,
                              hipStream_t stream) {
  const int*   y_t_1  = (const int*)  d_in[0];
  const float* h0     = (const float*)d_in[1];
  const float* c0     = (const float*)d_in[2];
  const float* c_t_1  = (const float*)d_in[3];
  const int*   ebv    = (const int*)  d_in[5];
  const float* cov    = (const float*)d_in[6];
  const float* emb    = (const float*)d_in[7];
  const float* Wx     = (const float*)d_in[8];
  const float* bx     = (const float*)d_in[9];
  const float* W_ih   = (const float*)d_in[10];
  const float* W_hh   = (const float*)d_in[11];
  const float* b_ih   = (const float*)d_in[12];
  const float* b_hh   = (const float*)d_in[13];
  const float* mem    = (const float*)d_in[14];
  const float* Wdp    = (const float*)d_in[15];
  const float* bdp    = (const float*)d_in[16];
  const float* v_w    = (const float*)d_in[17];
  const float* Wpg    = (const float*)d_in[18];
  const float* bpg    = (const float*)d_in[19];
  const float* Wo1    = (const float*)d_in[20];
  const float* bo1    = (const float*)d_in[21];
  const float* Wo2    = (const float*)d_in[22];
  const float* bo2    = (const float*)d_in[23];

  // output layout (flat, return order)
  float* o_fd   = (float*)d_out;
  float* o_h    = o_fd   + (size_t)B_ * VE_;
  float* o_c    = o_h    + B_ * H_;
  float* o_ct   = o_c    + B_ * H_;
  float* o_attn = o_ct   + B_ * 2 * H_;
  float* o_pg   = o_attn + B_ * TK_;
  float* o_cov  = o_pg   + B_;

  // workspace layout
  float* ws_x    = (float*)d_ws;                 // B*H
  float* ws_df   = ws_x + B_ * H_;               // B*2H
  f16*   ws_o1h  = (f16*)(ws_df + B_ * 2 * H_);  // B*H f16
  f16*   ws_w2h  = ws_o1h + B_ * H_;             // V*H f16 (~25.6 MB)

  // weight conversion: ~77 MB of L2/HBM traffic, ~3 us at 23.3 TB/s
  k_cvt_w<<<(V_ * H_) / (256 * 4), 256, 0, stream>>>(Wo2, ws_w2h, V_ * H_);

  k_proj_x<<<B_, 256, 0, stream>>>(c_t_1, emb, y_t_1, Wx, bx, ws_x);
  k_lstm  <<<B_, 256, 0, stream>>>(ws_x, h0, c0, W_ih, W_hh, b_ih, b_hh, o_h, o_c);
  k_decfea<<<B_, 256, 0, stream>>>(o_h, o_c, Wdp, bdp, ws_df);
  k_attn  <<<B_, 256, 0, stream>>>(ws_df, mem, v_w, o_attn, o_ct);
  k_out1_pgen<<<B_, 256, 0, stream>>>(o_h, o_c, o_ct, ws_x, Wpg, bpg, Wo1, bo1,
                                      o_pg, ws_o1h);

  dim3 ggrid(B_ / (16 * MTILES_PER_BLK), (3125 + 7) / 8);   // 8 x 391, 8 waves/block
  k_gemm_vocab<<<ggrid, 256, 0, stream>>>(ws_o1h, ws_w2h, bo2, o_fd);

  k_softmax<<<B_, 256, 0, stream>>>(o_fd, o_pg);
  k_scatter<<<(B_ * TK_) / 256, 256, 0, stream>>>(o_fd, o_attn, o_pg, ebv, cov, o_cov);
}
